// MinkUNet_25898652795318
// MI455X (gfx1250) — matrix operand order repaired and checked by reading, not compiled, not run on hardware
//
#include <hip/hip_runtime.h>
#include <hip/hip_bf16.h>

// ---------------------------------------------------------------------------
// MinkUNet-ish sparse conv network for MI455X (gfx1250, wave32, WMMA).
// Sparse convs run as gather + v_wmma_f32_16x16x32_f16 (f32 accumulate).
// Weights are pre-packed into the CDNA5 WMMA B-operand lane layout so each
// lane loads its v16h fragment with two b128 loads. Invalid neighbors are
// redirected to a zeroed row (single 64-bit pointer cndmask instead of 16
// per-register selects), keeping EXEC all-ones at every WMMA.
// ---------------------------------------------------------------------------

typedef __attribute__((ext_vector_type(16))) _Float16 v16h;
typedef __attribute__((ext_vector_type(8)))  float    v8f;
typedef __attribute__((ext_vector_type(4)))  unsigned int v4u;

union AFrag { v16h h; v4u q[2]; };

__device__ __forceinline__ unsigned short f2h(float f) {
    _Float16 h = (_Float16)f;
    return __builtin_bit_cast(unsigned short, h);
}

// ---------------------------------------------------------------------------
// Problem constants (fixed by the reference)
// ---------------------------------------------------------------------------
#define NPTS  120000
#define NPTS1 15000
#define K3    27
#define K2    8
#define CCH   32
#define NCLS  17

// ---------------------------------------------------------------------------
// Weight packing: W (K, Cin, 32) fp32 -> f16 tiles in WMMA B lane layout.
// Tile t = ((k*CinSlices + cs)*2 + nhalf), tile = [lane 0..31][elem 0..15]:
//   lane < 16 : element e holds W[k][cs*32 + (e<8 ? e : 16+e-8)][nhalf*16 + lane]
//   lane >=16 : element e holds W[k][cs*32 + (e<8 ? 8+e : 24+e-8)][nhalf*16 + (lane-16)]
// ---------------------------------------------------------------------------
__global__ void pack_w(const float* __restrict__ W, unsigned short* __restrict__ out,
                       int K, int Cin) {
    int CinSlices = Cin >> 5;
    int total = K * CinSlices * 2 * 512;
    for (int i = blockIdx.x * blockDim.x + threadIdx.x; i < total;
         i += gridDim.x * blockDim.x) {
        int e    = i & 15;
        int lane = (i >> 4) & 31;
        int tile = i >> 9;
        int nhalf = tile & 1;
        int rest  = tile >> 1;
        int cs    = rest % CinSlices;
        int k     = rest / CinSlices;
        int c = cs * 32 + ((e >> 3) << 4) + ((lane >> 4) << 3) + (e & 7);
        int n = nhalf * 16 + (lane & 15);
        out[i] = f2h(W[(k * Cin + c) * 32 + n]);
    }
}

// ---------------------------------------------------------------------------
// Gather-GEMM sparse conv via WMMA. One wave handles a 16-point tile and all
// 32 output channels (two 16x16 f32 accumulators). xh rows are f16 with
// stride `xstride` halves. zrow is a zero-filled row used for nbr == -1.
// ---------------------------------------------------------------------------
__global__ void conv16_wmma(const unsigned short* __restrict__ xh, int xstride,
                            const unsigned short* __restrict__ zrow,
                            const int* __restrict__ nbr, int K,
                            const unsigned short* __restrict__ wp, int CinSlices,
                            float* __restrict__ out, int M) {
    const int lane = threadIdx.x & 31;
    const int wave = threadIdx.x >> 5;
    const int tile = blockIdx.x * (blockDim.x >> 5) + wave;
    const int m0 = tile * 16;
    if (m0 >= M) return;                      // wave-uniform exit

    const int  row    = m0 + (lane & 15);     // A-operand row for this lane
    const bool rvalid = row < M;
    const int  ko     = (lane >> 4) << 3;     // 0 for lanes 0-15, 8 for 16-31

    v8f acc0 = {};
    v8f acc1 = {};

    for (int k = 0; k < K; ++k) {
        int  idx   = rvalid ? nbr[row * K + k] : -1;
        // Invalid neighbor -> zero row: one 64-bit address select, no register
        // masking, EXEC stays all-ones for the WMMAs.
        const unsigned short* xrow =
            (idx >= 0) ? (xh + (size_t)idx * xstride) : zrow;

        // Prefetch next k-offset's gather row to hide L2 latency behind WMMA.
        if (k + 1 < K && rvalid) {
            int idn = nbr[row * K + k + 1];
            if (idn >= 0) __builtin_prefetch(xh + (size_t)idn * xstride, 0, 0);
        }

        for (int cs = 0; cs < CinSlices; ++cs) {
            AFrag a;
            a.q[0] = *(const v4u*)(xrow + cs * 32 + ko);        // K = ko..ko+7
            a.q[1] = *(const v4u*)(xrow + cs * 32 + 16 + ko);   // K = 16+ko..

            const unsigned short* wt =
                wp + (((size_t)(k * CinSlices + cs) * 2) << 9);
            AFrag b0, b1;
            b0.q[0] = *(const v4u*)(wt + lane * 16);
            b0.q[1] = *(const v4u*)(wt + lane * 16 + 8);
            b1.q[0] = *(const v4u*)(wt + 512 + lane * 16);
            b1.q[1] = *(const v4u*)(wt + 512 + lane * 16 + 8);

            acc0 = __builtin_amdgcn_wmma_f32_16x16x32_f16(
                false, a.h, false, b0.h, (short)0, acc0, false, false);
            acc1 = __builtin_amdgcn_wmma_f32_16x16x32_f16(
                false, a.h, false, b1.h, (short)0, acc1, false, false);
        }
    }

    // C/D layout: VGPR r -> (M = r + (lane>=16 ? 8 : 0), N = lane & 15)
    const int ncol = lane & 15;
    const int mb   = m0 + ((lane >> 4) << 3);
#pragma unroll
    for (int r = 0; r < 8; ++r) {
        int mr = mb + r;
        if (mr < M) {
            out[(size_t)mr * 32 + ncol]      = acc0[r];
            out[(size_t)mr * 32 + 16 + ncol] = acc1[r];
        }
    }
}

// ---------------------------------------------------------------------------
// Stem conv (Cin = 3) — too thin for WMMA, plain VALU.
// ---------------------------------------------------------------------------
__global__ void stem0_conv(const float* __restrict__ x, const int* __restrict__ nbr,
                           const float* __restrict__ W, float* __restrict__ out, int M) {
    int t = blockIdx.x * blockDim.x + threadIdx.x;
    if (t >= M * 32) return;
    int m = t >> 5, o = t & 31;
    float acc = 0.f;
    for (int k = 0; k < K3; ++k) {
        int idx = nbr[m * K3 + k];
        if (idx >= 0) {
            const float* xr = x + (size_t)idx * 3;
            const float* wr = W + (size_t)(k * 3) * 32 + o;
            acc += xr[0] * wr[0] + xr[1] * wr[32] + xr[2] * wr[64];
        }
    }
    out[t] = acc;
}

// ---------------------------------------------------------------------------
// Per-channel sum / sum-of-squares (batch-norm statistics). stats[0..31]=sum,
// stats[32..63]=sumsq; caller zeroes stats first.
// ---------------------------------------------------------------------------
__global__ void chan_stats(const float* __restrict__ x, float* __restrict__ stats, int M) {
    __shared__ float ssum[32], ssq[32];
    int tid = threadIdx.x;
    if (tid < 32) { ssum[tid] = 0.f; ssq[tid] = 0.f; }
    __syncthreads();
    int c = tid & 31;                       // stride is a multiple of 32
    float s = 0.f, s2 = 0.f;
    for (int i = blockIdx.x * blockDim.x + tid; i < M * 32;
         i += gridDim.x * blockDim.x) {
        float v = x[i];
        s += v; s2 += v * v;
    }
    atomicAdd(&ssum[c], s);
    atomicAdd(&ssq[c], s2);
    __syncthreads();
    if (tid < 32) {
        atomicAdd(&stats[tid], ssum[tid]);
        atomicAdd(&stats[32 + tid], ssq[tid]);
    }
}

// ---------------------------------------------------------------------------
// Batch-norm apply (+ optional ReLU), writing optional f32 (strided) and f16
// (strided) copies for the downstream consumers.
// ---------------------------------------------------------------------------
__global__ void bn_apply(const float* __restrict__ x, const float* __restrict__ stats,
                         const float* __restrict__ gb, int do_relu,
                         float* __restrict__ outf, int fstride, int foff,
                         unsigned short* __restrict__ outh, int hstride, int hoff,
                         int M) {
    int i = blockIdx.x * blockDim.x + threadIdx.x;
    if (i >= M * 32) return;
    int c = i & 31, m = i >> 5;
    float mu  = stats[c] / (float)M;
    float var = stats[32 + c] / (float)M - mu * mu;
    float sc  = gb[c] * rsqrtf(var + 1e-5f);
    float v   = sc * (x[i] - mu) + gb[32 + c];
    if (do_relu) v = fmaxf(v, 0.f);
    if (outf) outf[(size_t)m * fstride + foff + c] = v;
    if (outh) outh[(size_t)m * hstride + hoff + c] = f2h(v);
}

// ---------------------------------------------------------------------------
// MobileViTv2 attention over all points (global softmax).
// ---------------------------------------------------------------------------
__global__ void att_logits(const float* __restrict__ net, const float* __restrict__ wi,
                           const float* __restrict__ bi, float* __restrict__ logits,
                           unsigned int* __restrict__ mxkey, int M) {
    int m = blockIdx.x * blockDim.x + threadIdx.x;
    if (m >= M) return;
    float s = bi[0];
    for (int j = 0; j < 32; ++j) s += net[(size_t)m * 32 + j] * wi[j];
    logits[m] = s;
    unsigned int u = __builtin_bit_cast(unsigned int, s);
    unsigned int key = (u & 0x80000000u) ? ~u : (u | 0x80000000u);
    atomicMax(mxkey, key);
}

__global__ void att_ctx(const float* __restrict__ net, const float* __restrict__ logits,
                        const float* __restrict__ wk, const float* __restrict__ bk,
                        const unsigned int* __restrict__ mxkey,
                        float* __restrict__ denom, float* __restrict__ ctx, int M) {
    __shared__ float sctx[32];
    __shared__ float sden;
    int tid = threadIdx.x;
    if (tid < 32) sctx[tid] = 0.f;
    if (tid == 0) sden = 0.f;
    __syncthreads();
    unsigned int key = *mxkey;
    unsigned int u = (key & 0x80000000u) ? (key ^ 0x80000000u) : ~key;
    float mx = __builtin_bit_cast(float, u);
    int c = tid & 31;
    float accC = 0.f, accD = 0.f;
    for (int i = blockIdx.x * blockDim.x + tid; i < M * 32;
         i += gridDim.x * blockDim.x) {
        int m = i >> 5;
        float e = __expf(logits[m] - mx);
        float y = bk[c];
        for (int j = 0; j < 32; ++j) y += net[(size_t)m * 32 + j] * wk[j * 32 + c];
        accC += e * y;
        if (c == 0) accD += e;
    }
    atomicAdd(&sctx[c], accC);
    if (c == 0) atomicAdd(&sden, accD);
    __syncthreads();
    if (tid < 32) atomicAdd(&ctx[tid], sctx[tid]);
    if (tid == 0) atomicAdd(denom, sden);
}

__global__ void att_v(const float* __restrict__ net, const float* __restrict__ wv,
                      const float* __restrict__ bv, const float* __restrict__ ctx,
                      const float* __restrict__ denom, float* __restrict__ v, int M) {
    int i = blockIdx.x * blockDim.x + threadIdx.x;
    if (i >= M * 32) return;
    int m = i >> 5, c = i & 31;
    float y = bv[c];
    for (int j = 0; j < 32; ++j) y += net[(size_t)m * 32 + j] * wv[j * 32 + c];
    v[i] = y * (ctx[c] / denom[0]);
}

// r = relu(v @ wo + bo + d), written as f16 for the up-conv gather input.
__global__ void att_out(const float* __restrict__ v, const float* __restrict__ wo,
                        const float* __restrict__ bo, const float* __restrict__ d,
                        unsigned short* __restrict__ r_h, int M) {
    int i = blockIdx.x * blockDim.x + threadIdx.x;
    if (i >= M * 32) return;
    int m = i >> 5, o = i & 31;
    float s = bo[o];
    for (int j = 0; j < 32; ++j) s += v[(size_t)m * 32 + j] * wo[j * 32 + o];
    s += d[i];
    r_h[i] = f2h(fmaxf(s, 0.f));
}

// ---------------------------------------------------------------------------
// 1x1 downsample shortcut: cat (M,64) @ W_ds (64,32)
// ---------------------------------------------------------------------------
__global__ void ds_mm(const float* __restrict__ cat, const float* __restrict__ Wds,
                      float* __restrict__ out, int M) {
    int i = blockIdx.x * blockDim.x + threadIdx.x;
    if (i >= M * 32) return;
    int m = i >> 5, o = i & 31;
    float s = 0.f;
    for (int j = 0; j < 64; ++j) s += cat[(size_t)m * 64 + j] * Wds[j * 32 + o];
    out[i] = s;
}

// y = relu(net2 + bn(ds));  out = y @ W_cls + b_cls
__global__ void final_cls(const float* __restrict__ net2, const float* __restrict__ dsx,
                          const float* __restrict__ stats, const float* __restrict__ gb,
                          const float* __restrict__ Wcls, const float* __restrict__ bcls,
                          float* __restrict__ out, int M) {
    int m = blockIdx.x * blockDim.x + threadIdx.x;
    if (m >= M) return;
    float y[32];
#pragma unroll
    for (int c = 0; c < 32; ++c) {
        float mu  = stats[c] / (float)M;
        float var = stats[32 + c] / (float)M - mu * mu;
        float sc  = gb[c] * rsqrtf(var + 1e-5f);
        float dv  = sc * (dsx[(size_t)m * 32 + c] - mu) + gb[32 + c];
        y[c] = fmaxf(net2[(size_t)m * 32 + c] + dv, 0.f);
    }
    for (int o = 0; o < NCLS; ++o) {
        float s = bcls[o];
#pragma unroll
        for (int c = 0; c < 32; ++c) s += y[c] * Wcls[c * NCLS + o];
        out[(size_t)m * NCLS + o] = s;
    }
}

// ---------------------------------------------------------------------------
// Host-side orchestration
// ---------------------------------------------------------------------------
static inline int cdiv(int a, int b) { return (a + b - 1) / b; }

extern "C" void kernel_launch(void* const* d_in, const int* in_sizes, int n_in,
                              void* d_out, int out_size, void* d_ws, size_t ws_size,
                              hipStream_t stream) {
    // -------- inputs (setup_inputs order) --------
    const float* x        = (const float*)d_in[0];
    const float* W_stem0  = (const float*)d_in[1];
    const float* bn_stem0 = (const float*)d_in[2];
    const float* W_stem1  = (const float*)d_in[3];
    const float* bn_stem1 = (const float*)d_in[4];
    const float* W_down   = (const float*)d_in[5];
    const float* bn_down  = (const float*)d_in[6];
    const float* W_res1   = (const float*)d_in[7];
    const float* bn_res1  = (const float*)d_in[8];
    const float* W_res2   = (const float*)d_in[9];
    const float* bn_res2  = (const float*)d_in[10];
    const float* att_wi   = (const float*)d_in[11];
    const float* att_bi   = (const float*)d_in[12];
    const float* att_wk   = (const float*)d_in[13];
    const float* att_bk   = (const float*)d_in[14];
    const float* att_wv   = (const float*)d_in[15];
    const float* att_bv   = (const float*)d_in[16];
    const float* att_wo   = (const float*)d_in[17];
    const float* att_bo   = (const float*)d_in[18];
    const float* W_up     = (const float*)d_in[19];
    const float* bn_up    = (const float*)d_in[20];
    const float* W_fuse1  = (const float*)d_in[21];
    const float* bn_fuse1 = (const float*)d_in[22];
    const float* W_fuse2  = (const float*)d_in[23];
    const float* bn_fuse2 = (const float*)d_in[24];
    const float* W_ds     = (const float*)d_in[25];
    const float* bn_ds    = (const float*)d_in[26];
    const float* W_cls    = (const float*)d_in[27];
    const float* b_cls    = (const float*)d_in[28];
    const int*   nbr0     = (const int*)d_in[29];
    const int*   nbr_down = (const int*)d_in[30];
    const int*   nbr1     = (const int*)d_in[31];
    const int*   nbr_up   = (const int*)d_in[32];
    float* out = (float*)d_out;

    // -------- workspace layout --------
    char* ws = (char*)d_ws;
    // packed weights (f16 halves, offsets in halves)
    unsigned short* wp = (unsigned short*)ws;
    const size_t p_stem1 = 0;                       // 27*2*512
    const size_t p_down  = p_stem1 + 27648;         //  8*2*512
    const size_t p_res1  = p_down  + 8192;
    const size_t p_res2  = p_res1  + 27648;
    const size_t p_up    = p_res2  + 27648;
    const size_t p_fuse1 = p_up    + 8192;          // 27*2*2*512
    const size_t p_fuse2 = p_fuse1 + 55296;
    // byte offsets for the rest
    const size_t o_stats = 393216;                  // 64 floats
    const size_t o_att   = 393728;                  // mxkey, denom, ctx[32]
    const size_t o_zero  = 394240;                  // 128B zero row (64 f16)
    const size_t o_conv  = 458752;                  // f32 (N,32) conv scratch
    const size_t o_ha    = o_conv + (size_t)NPTS * 32 * 4;   // f16 (N,32) scratch
    const size_t o_catf  = o_ha   + (size_t)NPTS * 32 * 2;   // f32 (N,64)
    const size_t o_cath  = o_catf + (size_t)NPTS * 64 * 4;   // f16 (N,64)
    const size_t o_net2  = o_cath + (size_t)NPTS * 64 * 2;   // f32 (N,32)
    const size_t o_df    = o_net2 + (size_t)NPTS * 32 * 4;   // f32 (N1,32)
    const size_t o_dh    = o_df   + (size_t)NPTS1 * 32 * 4;  // f16 (N1,32)
    const size_t o_net1  = o_dh   + (size_t)NPTS1 * 32 * 2;  // f32 (N1,32)
    const size_t o_log   = o_net1 + (size_t)NPTS1 * 32 * 4;  // f32 (N1)
    const size_t o_v     = o_log  + (size_t)NPTS1 * 4;       // f32 (N1,32)

    float*          stats = (float*)(ws + o_stats);
    unsigned int*   mxkey = (unsigned int*)(ws + o_att);
    float*          denom = (float*)(ws + o_att + 4);
    float*          ctx   = (float*)(ws + o_att + 8);
    unsigned short* zrow  = (unsigned short*)(ws + o_zero);
    float*          conv  = (float*)(ws + o_conv);
    unsigned short* ha    = (unsigned short*)(ws + o_ha);
    float*          catf  = (float*)(ws + o_catf);
    unsigned short* cath  = (unsigned short*)(ws + o_cath);
    float*          net2  = (float*)(ws + o_net2);
    float*          df    = (float*)(ws + o_df);
    unsigned short* dh    = (unsigned short*)(ws + o_dh);
    float*          net1  = (float*)(ws + o_net1);
    float*          logit = (float*)(ws + o_log);
    float*          vbuf  = (float*)(ws + o_v);

    const int TB = 256;
    const int gN32  = cdiv(NPTS * 32, TB);
    const int gN132 = cdiv(NPTS1 * 32, TB);
    const dim3 b(TB);

#define STATS(buf, M)                                                        \
    hipMemsetAsync(stats, 0, 64 * sizeof(float), stream);                    \
    chan_stats<<<512, b, 0, stream>>>(buf, stats, M)

    // zero row for invalid-neighbor redirect
    hipMemsetAsync(zrow, 0, 128, stream);

    // -------- pack all conv weights into WMMA B layout --------
    pack_w<<<64, b, 0, stream>>>(W_stem1, wp + p_stem1, K3, 32);
    pack_w<<<64, b, 0, stream>>>(W_down,  wp + p_down,  K2, 32);
    pack_w<<<64, b, 0, stream>>>(W_res1,  wp + p_res1,  K3, 32);
    pack_w<<<64, b, 0, stream>>>(W_res2,  wp + p_res2,  K3, 32);
    pack_w<<<64, b, 0, stream>>>(W_up,    wp + p_up,    K2, 32);
    pack_w<<<128, b, 0, stream>>>(W_fuse1, wp + p_fuse1, K3, 64);
    pack_w<<<64, b, 0, stream>>>(W_fuse2, wp + p_fuse2, K3, 32);

    // -------- stem --------
    stem0_conv<<<gN32, b, 0, stream>>>(x, nbr0, W_stem0, conv, NPTS);
    STATS(conv, NPTS);
    bn_apply<<<gN32, b, 0, stream>>>(conv, stats, bn_stem0, 1,
                                     nullptr, 0, 0, ha, 32, 0, NPTS);

    conv16_wmma<<<cdiv(cdiv(NPTS, 16), 8), b, 0, stream>>>(
        ha, 32, zrow, nbr0, K3, wp + p_stem1, 1, conv, NPTS);
    STATS(conv, NPTS);
    // x0 -> cat columns 32..63 (f32 + f16)
    bn_apply<<<gN32, b, 0, stream>>>(conv, stats, bn_stem1, 1,
                                     catf, 64, 32, cath, 64, 32, NPTS);

    // -------- downsample --------
    conv16_wmma<<<cdiv(cdiv(NPTS1, 16), 8), b, 0, stream>>>(
        cath + 32, 64, zrow, nbr_down, K2, wp + p_down, 1, conv, NPTS1);
    STATS(conv, NPTS1);
    bn_apply<<<gN132, b, 0, stream>>>(conv, stats, bn_down, 1,
                                      df, 32, 0, dh, 32, 0, NPTS1);

    // -------- residual block with attention --------
    conv16_wmma<<<cdiv(cdiv(NPTS1, 16), 8), b, 0, stream>>>(
        dh, 32, zrow, nbr1, K3, wp + p_res1, 1, conv, NPTS1);
    STATS(conv, NPTS1);
    bn_apply<<<gN132, b, 0, stream>>>(conv, stats, bn_res1, 1,
                                      nullptr, 0, 0, ha, 32, 0, NPTS1);

    conv16_wmma<<<cdiv(cdiv(NPTS1, 16), 8), b, 0, stream>>>(
        ha, 32, zrow, nbr1, K3, wp + p_res2, 1, conv, NPTS1);
    STATS(conv, NPTS1);
    bn_apply<<<gN132, b, 0, stream>>>(conv, stats, bn_res2, 0,
                                      net1, 32, 0, nullptr, 0, 0, NPTS1);

    hipMemsetAsync((void*)mxkey, 0, 512, stream);  // mxkey, denom, ctx
    att_logits<<<cdiv(NPTS1, TB), b, 0, stream>>>(net1, att_wi, att_bi,
                                                  logit, mxkey, NPTS1);
    att_ctx<<<256, b, 0, stream>>>(net1, logit, att_wk, att_bk, mxkey,
                                   denom, ctx, NPTS1);
    att_v<<<gN132, b, 0, stream>>>(net1, att_wv, att_bv, ctx, denom, vbuf, NPTS1);
    att_out<<<gN132, b, 0, stream>>>(vbuf, att_wo, att_bo, df, ha, NPTS1);

    // -------- upsample --------
    conv16_wmma<<<cdiv(cdiv(NPTS, 16), 8), b, 0, stream>>>(
        ha, 32, zrow, nbr_up, K2, wp + p_up, 1, conv, NPTS);
    STATS(conv, NPTS);
    // u -> cat columns 0..31 (f32 + f16)
    bn_apply<<<gN32, b, 0, stream>>>(conv, stats, bn_up, 1,
                                     catf, 64, 0, cath, 64, 0, NPTS);

    // -------- fuse residual block (2C -> C) --------
    conv16_wmma<<<cdiv(cdiv(NPTS, 16), 8), b, 0, stream>>>(
        cath, 64, zrow, nbr0, K3, wp + p_fuse1, 2, conv, NPTS);
    STATS(conv, NPTS);
    bn_apply<<<gN32, b, 0, stream>>>(conv, stats, bn_fuse1, 1,
                                     nullptr, 0, 0, ha, 32, 0, NPTS);

    conv16_wmma<<<cdiv(cdiv(NPTS, 16), 8), b, 0, stream>>>(
        ha, 32, zrow, nbr0, K3, wp + p_fuse2, 1, conv, NPTS);
    STATS(conv, NPTS);
    bn_apply<<<gN32, b, 0, stream>>>(conv, stats, bn_fuse2, 0,
                                     net2, 32, 0, nullptr, 0, 0, NPTS);

    // -------- shortcut + classifier --------
    ds_mm<<<gN32, b, 0, stream>>>(catf, W_ds, conv, NPTS);
    STATS(conv, NPTS);
    final_cls<<<cdiv(NPTS, TB), b, 0, stream>>>(net2, conv, stats, bn_ds,
                                                W_cls, b_cls, out, NPTS);
#undef STATS
    (void)n_in; (void)in_sizes; (void)out_size; (void)ws_size;
}
